// SpatialCrossAttention_58317065945163
// MI455X (gfx1250) — compile-verified, matrix operand order checked by reading, not run on hardware
//
#include <hip/hip_runtime.h>

// ---------------------------------------------------------------------------
// Spatial cross attention (BEVFormer-style) for gfx1250 / MI455X.
// EMBED=256, HEADS=8, HEAD_DIM=32, LEVELS=4, POINTS=8, CAMS=6, ZANCH=4,
// NQ=10000, NV=19560, BS=1.
// ---------------------------------------------------------------------------

#define EMBED 256
#define HEADS 8
#define HEAD_DIM 32
#define LEVELS 4
#define POINTS 8
#define CAMS 6
#define ZANCH 4
#define NQ 10000
#define NV 19560

typedef __attribute__((ext_vector_type(16))) __bf16 v16bf;
typedef __attribute__((ext_vector_type(8)))  __bf16 v8bf;
typedef __attribute__((ext_vector_type(8)))  float  v8f;

// LDS weight tile: 8 k-blocks x 64 columns, each (kblock,col) holds 32 bf16
// K-values pre-swizzled into WMMA B-fragment order [K0..7,K16..23,K8..15,K24..31],
// padded to stride 40 elements (80 B) for conflict-free ds_load_b128 reads.
#define LDS_STRIDE 40

// ---------------------------------------------------------------------------
// bf16 WMMA GEMM: out[M,N] = (A (+A2)) @ W[K=256,N] + bias (+resid)
// Block = 128 threads = 4 waves. Each wave computes a 16x64 output strip
// (4 accumulators) over K=256 -> 32 v_wmma_f32_16x16x32_bf16 per wave.
// Grid: (ceil(M/64), N/64).
// ---------------------------------------------------------------------------
template <bool HAS_A2, bool OUT_BF16>
__global__ __launch_bounds__(128)
void gemm_wmma(const float* __restrict__ A, const float* __restrict__ A2,
               const float* __restrict__ W, const float* __restrict__ bias,
               const float* __restrict__ resid,
               float* __restrict__ outF, __bf16* __restrict__ outH,
               int M, int N) {
  __shared__ __bf16 ldsW[8 * 64 * LDS_STRIDE];   // 40 KB

  const int tid   = threadIdx.x;
  const int wave  = tid >> 5;
  const int lane  = tid & 31;
  const int nBase = blockIdx.y * 64;

  // Cooperative load of the 256 x 64 fp32 weight tile -> swizzled bf16 LDS.
  // pos permutation: K%32 -> [0..7 ->0..7, 8..15 ->16..23, 16..23 ->8..15, 24..31 ->24..31]
  for (int idx = tid; idx < 256 * 64; idx += 128) {
    const int k  = idx >> 6;          // 0..255
    const int n  = idx & 63;
    const int kk = k >> 5;
    const int k5 = k & 31;
    const int pos = (k5 & 7) | ((k5 & 8) << 1) | ((k5 & 16) >> 1);
    ldsW[(kk * 64 + n) * LDS_STRIDE + pos] = (__bf16)W[(size_t)k * N + nBase + n];
  }
  __syncthreads();

  const int  m15   = lane & 15;
  const int  half  = lane >> 4;          // 0: K {0..7,16..23}; 1: K {8..15,24..31}
  const int  khalf = half * 8;
  const int  tileM = blockIdx.x * 4 + wave;
  const bool active = (tileM * 16) < M;
  const int  row   = tileM * 16 + m15;

  const float* ap  = A + (size_t)(active ? row : 0) * 256;
  const float* ap2 = HAS_A2 ? (A2 + (size_t)(active ? row : 0) * 256) : nullptr;

  v8f c0 = {}, c1 = {}, c2 = {}, c3 = {};

#pragma unroll
  for (int kk = 0; kk < 8; ++kk) {
    const int kb = kk * 32;
    // A fragment: two contiguous 8-float segments, fp32 -> bf16.
    v16bf a;
#pragma unroll
    for (int j = 0; j < 8; ++j) {
      float lo = ap[kb + khalf + j];
      float hi = ap[kb + 16 + khalf + j];
      if (HAS_A2) {
        lo += ap2[kb + khalf + j];
        hi += ap2[kb + 16 + khalf + j];
      }
      a[j]     = (__bf16)lo;
      a[8 + j] = (__bf16)hi;
    }
    // 4 B fragments from swizzled LDS; two 16 B ds loads each.
#pragma unroll
    for (int sub = 0; sub < 4; ++sub) {
      const int ebase = (kk * 64 + sub * 16 + m15) * LDS_STRIDE + half * 16;
      const v8bf blo = *(const v8bf*)&ldsW[ebase];
      const v8bf bhi = *(const v8bf*)&ldsW[ebase + 8];
      v16bf b;
#pragma unroll
      for (int j = 0; j < 8; ++j) { b[j] = blo[j]; b[8 + j] = bhi[j]; }
      v8f& c = (sub == 0) ? c0 : (sub == 1) ? c1 : (sub == 2) ? c2 : c3;
      c = __builtin_amdgcn_wmma_f32_16x16x32_bf16(
          false, a, false, b, (short)0, c, false, false);
    }
  }

  if (!active) return;

  // Epilogue. D layout: VGPR i -> M = i + 8*(lane>>4); N = lane&15.
#pragma unroll
  for (int sub = 0; sub < 4; ++sub) {
    const v8f& c = (sub == 0) ? c0 : (sub == 1) ? c1 : (sub == 2) ? c2 : c3;
    const int colg = nBase + sub * 16 + m15;
    const float bb = bias ? bias[colg] : 0.f;
#pragma unroll
    for (int i = 0; i < 8; ++i) {
      const int rowg = tileM * 16 + i + 8 * half;
      const size_t o = (size_t)rowg * N + colg;
      float val = c[i] + bb;
      if (resid) val += resid[o];
      if (OUT_BF16) outH[o] = (__bf16)val;
      else          outF[o] = val;
    }
  }
}

// ---------------------------------------------------------------------------
// Wave-level softmax over 32 contiguous logits per (query, head).
// ---------------------------------------------------------------------------
__global__ __launch_bounds__(128)
void softmax32_kernel(float* __restrict__ att) {
  const int gw   = (blockIdx.x * blockDim.x + threadIdx.x) >> 5;
  const int lane = threadIdx.x & 31;
  if (gw >= NQ * HEADS) return;
  float v = att[(size_t)gw * 32 + lane];
  float m = v;
#pragma unroll
  for (int s = 16; s > 0; s >>= 1) m = fmaxf(m, __shfl_xor(m, s, 32));
  const float e = __expf(v - m);
  float sum = e;
#pragma unroll
  for (int s = 16; s > 0; s >>= 1) sum += __shfl_xor(sum, s, 32);
  att[(size_t)gw * 32 + lane] = e / sum;
}

// ---------------------------------------------------------------------------
// Deformable sampling. One wave per (query, head); lane = head-dim channel so
// every bilinear-corner fetch is a contiguous 64 B bf16 read (L2-resident).
// All per-sample scalars (coords, weights, masks) are wave-uniform.
// ---------------------------------------------------------------------------
__global__ __launch_bounds__(128)
void sample_kernel(const float* __restrict__ off, const float* __restrict__ attw,
                   const __bf16* __restrict__ vproj, const float* __restrict__ ref,
                   const unsigned char* __restrict__ mask, float* __restrict__ slots) {
  const int gw   = (blockIdx.x * blockDim.x + threadIdx.x) >> 5;
  const int lane = threadIdx.x & 31;
  if (gw >= NQ * HEADS) return;
  const int q = gw >> 3;
  const int h = gw & 7;

  const int LH[LEVELS] = {92, 46, 23, 12};
  const int LW[LEVELS] = {160, 80, 40, 20};
  const int LS[LEVELS] = {0, 14720, 18400, 19320};

  float acc = 0.f;
  int   count = 0;

  for (int cam = 0; cam < CAMS; ++cam) {
    // bev_mask[cam, 0, q, 0..3]: 4 bool bytes -> any() via one u32 load.
    const unsigned int mb =
        *(const unsigned int*)(mask + ((size_t)cam * NQ + q) * ZANCH);
    if (!mb) continue;
    ++count;

    const float* refq  = ref + ((size_t)cam * NQ + q) * ZANCH * 2;
    const size_t vbase = (size_t)cam * NV;

    for (int l = 0; l < LEVELS; ++l) {
      const int   Wl = LW[l], Hl = LH[l];
      const float fW = (float)Wl, fH = (float)Hl;
      const size_t lvbase = vbase + LS[l];
#pragma unroll
      for (int p = 0; p < POINTS; ++p) {
        const float aw = attw[(size_t)q * 256 + h * 32 + l * 8 + p];
        const int oidx = ((h * LEVELS + l) * POINTS + p) * 2;
        const float ox = off[(size_t)q * 512 + oidx];
        const float oy = off[(size_t)q * 512 + oidx + 1];
        const float rx = refq[(p & 3) * 2];
        const float ry = refq[(p & 3) * 2 + 1];
        // loc = ref + off/norm; x = loc.x*W - 0.5  =>  ref.x*W + off.x - 0.5
        const float x = rx * fW + ox - 0.5f;
        const float y = ry * fH + oy - 0.5f;
        const float x0 = floorf(x), y0 = floorf(y);
        const float lx = x - x0, ly = y - y0;
        const int xi = (int)x0, yi = (int)y0;
#pragma unroll
        for (int cr = 0; cr < 4; ++cr) {
          const int dx = cr & 1, dy = cr >> 1;
          const int xx = xi + dx, yy = yi + dy;
          if (xx < 0 || xx >= Wl || yy < 0 || yy >= Hl) continue;
          const float cw = (dx ? lx : 1.f - lx) * (dy ? ly : 1.f - ly);
          const float v =
              (float)vproj[(lvbase + (size_t)yy * Wl + xx) * EMBED + h * 32 + lane];
          acc += aw * cw * v;
        }
      }
    }
  }
  acc /= (float)(count < 1 ? 1 : count);
  slots[(size_t)q * 256 + h * 32 + lane] = acc;
}

// ---------------------------------------------------------------------------
// Host launch.
// ---------------------------------------------------------------------------
extern "C" void kernel_launch(void* const* d_in, const int* in_sizes, int n_in,
                              void* d_out, int out_size, void* d_ws, size_t ws_size,
                              hipStream_t stream) {
  (void)in_sizes; (void)n_in; (void)out_size; (void)ws_size;
  const float*         query     = (const float*)d_in[0];
  const float*         query_pos = (const float*)d_in[1];
  const float*         value     = (const float*)d_in[2];   // (CAMS, NV, 1, 256)
  const float*         refpts    = (const float*)d_in[3];   // (CAMS, 1, NQ, 4, 2)
  const unsigned char* bev_mask  = (const unsigned char*)d_in[4]; // bool bytes
  // d_in[5]=spatial_shapes, d_in[6]=level_start_index: compile-time constants.
  const float* Wv   = (const float*)d_in[7];
  const float* bv   = (const float*)d_in[8];
  const float* Woff = (const float*)d_in[9];
  const float* boff = (const float*)d_in[10];
  const float* Watt = (const float*)d_in[11];
  const float* batt = (const float*)d_in[12];
  const float* Wout = (const float*)d_in[13];
  const float* bout = (const float*)d_in[14];
  float* out = (float*)d_out;

  // Workspace layout (256 B aligned offsets).
  char*   ws    = (char*)d_ws;
  float*  off   = (float*)(ws);                       // NQ*512*4  = 20,480,000 B
  float*  attw  = (float*)(ws + 20480000);            // NQ*256*4  = 10,240,000 B
  __bf16* vproj = (__bf16*)(ws + 30720000);           // 6*NV*256*2 = 60,088,320 B
  float*  slots = (float*)(ws + 90808320);            // NQ*256*4  = 10,240,000 B

  const dim3 blk(128);
  const int  mtilesQ = NQ / 16;                // 625
  const int  mtilesV = (CAMS * NV) / 16;       // 7335
  const int  gxQ = (mtilesQ + 3) / 4;          // 157
  const int  gxV = (mtilesV + 3) / 4;          // 1834

  // 1) Offsets: (q+qpos) @ Woff + boff -> off  (M=10000, N=512)
  gemm_wmma<true, false><<<dim3(gxQ, 512 / 64), blk, 0, stream>>>(
      query, query_pos, Woff, boff, nullptr, off, nullptr, NQ, 512);

  // 2) Attention logits: (q+qpos) @ Watt + batt -> attw (M=10000, N=256)
  gemm_wmma<true, false><<<dim3(gxQ, 256 / 64), blk, 0, stream>>>(
      query, query_pos, Watt, batt, nullptr, attw, nullptr, NQ, 256);

  // 3) Softmax over 32 (level,point) logits per head.
  softmax32_kernel<<<dim3((NQ * HEADS) / 4), blk, 0, stream>>>(attw);

  // 4) Value projection, all cameras at once: M = 6*19560 = 117360.
  gemm_wmma<false, true><<<dim3(gxV, 256 / 64), blk, 0, stream>>>(
      value, nullptr, Wv, bv, nullptr, nullptr, vproj, CAMS * NV, 256);

  // 5) Deformable bilinear sampling + camera mask/normalize -> slots.
  sample_kernel<<<dim3((NQ * HEADS) / 4), blk, 0, stream>>>(
      off, attw, vproj, refpts, bev_mask, slots);

  // 6) Output projection with residual: slots @ Wout + bout + query -> out.
  gemm_wmma<false, false><<<dim3(gxQ, 256 / 64), blk, 0, stream>>>(
      slots, nullptr, Wout, bout, query, out, nullptr, NQ, 256);
}
